// VarRNNBase_88656714924168
// MI455X (gfx1250) — matrix operand order repaired and checked by reading, not compiled, not run on hardware
//
#include <hip/hip_runtime.h>
#include <hip/hip_bf16.h>

// ---------------------------------------------------------------------------
// 2-layer bidirectional variational LSTM for MI455X (gfx1250).
//   S=512, B=64, I=512, H=512, 4H=2048.
// Strategy:
//   * xp = (x*mask_x) @ W_ih^T + b_ih + b_hh : big parallel bf16 WMMA GEMM.
//   * recurrent scan: persistent kernel, c-state in registers, h shared via
//     L2-resident ping-pong buffer, W_hh slice staged ONCE into 128 KB LDS
//     (CDNA5: 320 KB/WGP), grid sync (16 WGs/direction) per step.
//   * all matrix math via V_WMMA_F32_16X16X32_BF16 (f32 accumulate).
// ---------------------------------------------------------------------------

typedef __bf16 bf16_t;
typedef __attribute__((ext_vector_type(16))) __bf16 v16bf;
typedef __attribute__((ext_vector_type(8)))  __bf16 v8bf;
typedef __attribute__((ext_vector_type(8)))  float  v8f;

#define SEQ   512
#define BATCH 64
#define HID   512
#define GDIM  2048              // 4*H
#define MROWS (SEQ * BATCH)     // 32768

// -------------------------------------------------------------------- cvt --
__global__ void cvt_f32_bf16(const float* __restrict__ src,
                             bf16_t* __restrict__ dst, int n) {
  int i = blockIdx.x * blockDim.x + threadIdx.x;
  if (i < n) dst[i] = (bf16_t)src[i];
}

// ----------------------------------------------------------------- xp gemm --
// C[d, m, g] = sum_k (A[m,k]*mask[d, m%64, k])_bf16 * W[d,g,k]_bf16
//              + b_ih[2*layer+d, g] + b_hh[2*layer+d, g]
// One wave computes a 16x64 tile (4 WMMA accumulators), K in chunks of 32.
__global__ __launch_bounds__(256)
void xp_gemm(const float* __restrict__ Ain,    // [MROWS, K]
             const float* __restrict__ maskx,  // [2, BATCH, K]
             const bf16_t* __restrict__ Wb,    // [2, GDIM, K] bf16
             const float* __restrict__ b_ih,   // [4, GDIM]
             const float* __restrict__ b_hh,   // [4, GDIM]
             float* __restrict__ xp,           // [2, MROWS, GDIM]
             int K, int layer) {
  const int lane = threadIdx.x & 31;
  const int wave = (blockIdx.x * blockDim.x + threadIdx.x) >> 5;
  const int NG = GDIM / 64;          // 32 n-groups
  const int MT = MROWS / 16;         // 2048 m-tiles
  int t = wave;
  const int d = t / (MT * NG);  t -= d * (MT * NG);
  const int mt = t / NG;        const int ng = t - mt * NG;
  if (d >= 2) return;

  const int l15 = lane & 15, hi = lane >> 4;
  const int arow = mt * 16 + l15;        // global M row for A operand
  const int brow = arow & (BATCH - 1);   // batch index for dropout mask
  const float* xr = Ain + (size_t)arow * K;
  const float* mr = maskx + ((size_t)d * BATCH + brow) * K;

  v8f acc[4] = {};
  const int chunks = K >> 5;
  for (int kc = 0; kc < chunks; ++kc) {
    const int o0 = kc * 32 + hi * 8;   // first 8-elem run
    const int o1 = o0 + 16;            // second 8-elem run
    v8f x0 = *(const v8f*)(xr + o0);
    v8f x1 = *(const v8f*)(xr + o1);
    v8f m0 = *(const v8f*)(mr + o0);
    v8f m1 = *(const v8f*)(mr + o1);
    union { v16bf v; __bf16 e[16]; } a;
#pragma unroll
    for (int e = 0; e < 8; ++e) {
      a.e[e]     = (__bf16)(x0[e] * m0[e]);
      a.e[8 + e] = (__bf16)(x1[e] * m1[e]);
    }
#pragma unroll
    for (int q = 0; q < 4; ++q) {
      const int n = ng * 64 + q * 16 + l15;
      union { v16bf v; v8bf h[2]; } bmu;
      bmu.h[0] = *(const v8bf*)(Wb + ((size_t)d * GDIM + n) * K + kc * 32 + hi * 8);
      bmu.h[1] = *(const v8bf*)(Wb + ((size_t)d * GDIM + n) * K + kc * 32 + 16 + hi * 8);
      v16bf bm = bmu.v;
      acc[q] = __builtin_amdgcn_wmma_f32_16x16x32_bf16(
          false, a.v, false, bm, (short)0, acc[q], false, false);
    }
  }

  const int idx = 2 * layer + d;
#pragma unroll
  for (int q = 0; q < 4; ++q) {
    const int n = ng * 64 + q * 16 + l15;
    const float bias = b_ih[idx * GDIM + n] + b_hh[idx * GDIM + n];
    float* op = xp + ((size_t)d * MROWS + mt * 16) * GDIM + n;
#pragma unroll
    for (int r = 0; r < 8; ++r)
      op[(size_t)(r + 8 * hi) * GDIM] = acc[q][r] + bias;
  }
}

// --------------------------------------------------------------- lstm scan --
// Persistent kernel. 32 WGs x 256 thr: WGs [0,16) = forward, [16,32) = back.
// WG `wg` owns j-tiles {2wg, 2wg+1}; its 8 waves map to (mt 0..3) x (jl 0..1).
// The WG's W_hh slice (2 j-tiles x 4 gates x 16 cols x 512 K bf16 = 128 KB)
// is staged into LDS once; per step B operands come from ds_load_b128.
// Each wave keeps its c tile in registers, writes h (f32) to the output and
// h*mask_h (bf16) to a ping-pong buffer read as the next step's WMMA A
// operand. Grid sync per step via a monotonic per-direction atomic counter.
__global__ __launch_bounds__(256)
void lstm_scan(const float* __restrict__ xp,      // [2, MROWS, GDIM] (+biases)
               const bf16_t* __restrict__ Whh,    // [4, GDIM, HID] bf16
               const float* __restrict__ mask_h,  // [4, BATCH, HID]
               float* __restrict__ outp,          // [SEQ, BATCH, 2*HID]
               float* __restrict__ hn,            // [4, BATCH, HID]
               float* __restrict__ cn,            // [4, BATCH, HID]
               bf16_t* __restrict__ hm,           // [2 dirs][2][BATCH, HID]
               unsigned* __restrict__ counters,   // one per direction
               int layer) {
  extern __shared__ bf16_t wlds[];                // [128 rows][512 K] = 128 KB
  const int dir  = blockIdx.x >> 4;
  const int wg   = blockIdx.x & 15;
  const int lane = threadIdx.x & 31;
  const int wvin = threadIdx.x >> 5;              // 0..7
  const int mt = wvin >> 1;                       // batch tile 0..3
  const int jl = wvin & 1;                        // local j-tile 0..1
  const int jt = wg * 2 + jl;                     // global j-tile 0..31
  const int l15 = lane & 15, hi = lane >> 4;
  const int idx = 2 * layer + dir;
  const int m0 = mt * 16;
  const int col = jt * 16 + l15;
  bf16_t* hmd = hm + (size_t)dir * 2 * BATCH * HID;
  unsigned* cnt = counters + dir;

  // ---- stage this WG's W_hh slice into LDS (once) ----
  // LDS row r = (jl*4 + q)*16 + col_in_tile, each row = 512 K bf16.
  for (int it = 0; it < 32; ++it) {
    const int chunk = it * 256 + threadIdx.x;     // 8192 chunks of 8 bf16
    const int kpart = chunk & 63;
    const int rowi  = chunk >> 6;                 // 0..127
    const int ccol  = rowi & 15;
    const int q     = (rowi >> 4) & 3;
    const int jli   = rowi >> 6;
    const int n     = q * HID + (wg * 2 + jli) * 16 + ccol;
    const bf16_t* src = Whh + ((size_t)idx * GDIM + n) * HID + kpart * 8;
    *(v8bf*)(wlds + (size_t)rowi * HID + kpart * 8) = *(const v8bf*)src;
  }
  __syncthreads();

  float mh[8];
#pragma unroll
  for (int r = 0; r < 8; ++r)
    mh[r] = mask_h[((size_t)idx * BATCH + (m0 + r + 8 * hi)) * HID + col];
  const bf16_t* wrow[4];
#pragma unroll
  for (int q = 0; q < 4; ++q)
    wrow[q] = wlds + (size_t)((jl * 4 + q) * 16 + l15) * HID;

  v8f c = {};
  for (int t = 0; t < SEQ; ++t) {
    const int s = dir ? (SEQ - 1 - t) : t;
    const float* xpb = xp + ((size_t)dir * MROWS + (size_t)s * BATCH + m0) * GDIM;
    v8f acc[4] = {};
    if (t > 0) {
      // warm the xp gate rows for this step while the WMMA loop runs
      {
        const size_t ro = (size_t)(8 * hi) * GDIM;
#pragma unroll
        for (int q = 0; q < 4; ++q)
          __builtin_prefetch(xpb + ro + q * HID + col, 0, 3);
      }
      const bf16_t* hr = hmd + (size_t)((t - 1) & 1) * BATCH * HID
                             + (size_t)(m0 + l15) * HID;
#pragma unroll 4
      for (int kc = 0; kc < HID / 32; ++kc) {
        union { v16bf v; v8bf h[2]; } a;
        a.h[0] = *(const v8bf*)(hr + kc * 32 + hi * 8);
        a.h[1] = *(const v8bf*)(hr + kc * 32 + 16 + hi * 8);
#pragma unroll
        for (int q = 0; q < 4; ++q) {
          union { v16bf v; v8bf h[2]; } bm;
          bm.h[0] = *(const v8bf*)(wrow[q] + kc * 32 + hi * 8);
          bm.h[1] = *(const v8bf*)(wrow[q] + kc * 32 + 16 + hi * 8);
          acc[q] = __builtin_amdgcn_wmma_f32_16x16x32_bf16(
              false, a.v, false, bm.v, (short)0, acc[q], false, false);
        }
      }
    }
    float hval[8];
#pragma unroll
    for (int r = 0; r < 8; ++r) {
      const size_t ro = (size_t)(r + 8 * hi) * GDIM;
      const float gi = acc[0][r] + xpb[ro + 0 * HID + col];
      const float gf = acc[1][r] + xpb[ro + 1 * HID + col];
      const float gg = acc[2][r] + xpb[ro + 2 * HID + col];
      const float go = acc[3][r] + xpb[ro + 3 * HID + col];
      const float si = 1.f / (1.f + __expf(-gi));
      const float sf = 1.f / (1.f + __expf(-gf));
      const float so = 1.f / (1.f + __expf(-go));
      const float tg = 2.f / (1.f + __expf(-2.f * gg)) - 1.f;
      const float cv = sf * c[r] + si * tg;
      c[r] = cv;
      const float tc = 2.f / (1.f + __expf(-2.f * cv)) - 1.f;
      hval[r] = so * tc;
    }
    bf16_t* hw = hmd + (size_t)(t & 1) * BATCH * HID;
#pragma unroll
    for (int r = 0; r < 8; ++r) {
      const int b = m0 + r + 8 * hi;
      outp[((size_t)s * BATCH + b) * (2 * HID) + dir * HID + col] = hval[r];
      hw[(size_t)b * HID + col] = (bf16_t)(hval[r] * mh[r]);
    }
    if (t == SEQ - 1) {
#pragma unroll
      for (int r = 0; r < 8; ++r) {
        const int b = m0 + r + 8 * hi;
        hn[((size_t)idx * BATCH + b) * HID + col] = hval[r];
        cn[((size_t)idx * BATCH + b) * HID + col] = c[r];
      }
    }
    // ---- grid sync within this direction (16 workgroups) ----
    __threadfence();
    __syncthreads();
    if (threadIdx.x == 0) {
      __hip_atomic_fetch_add(cnt, 1u, __ATOMIC_RELEASE, __HIP_MEMORY_SCOPE_AGENT);
      const unsigned target = 16u * (unsigned)(t + 1);
      while (__hip_atomic_load(cnt, __ATOMIC_ACQUIRE, __HIP_MEMORY_SCOPE_AGENT) < target)
        __builtin_amdgcn_s_sleep(1);
    }
    __syncthreads();
  }
}

// ------------------------------------------------------------------ launch --
extern "C" void kernel_launch(void* const* d_in, const int* in_sizes, int n_in,
                              void* d_out, int out_size, void* d_ws, size_t ws_size,
                              hipStream_t stream) {
  const float* x     = (const float*)d_in[0];
  const float* Wih0  = (const float*)d_in[1];  // [2, 2048, 512]
  const float* Wih1  = (const float*)d_in[2];  // [2, 2048, 1024]
  const float* Whh   = (const float*)d_in[3];  // [4, 2048, 512]
  const float* b_ih  = (const float*)d_in[4];
  const float* b_hh  = (const float*)d_in[5];
  const float* mx0   = (const float*)d_in[6];
  const float* mx1   = (const float*)d_in[7];
  const float* mhh   = (const float*)d_in[8];

  // d_out = [S,B,2H] out ++ [4,B,H] hn ++ [4,B,H] cn
  float* out_main = (float*)d_out;
  float* hn = out_main + (size_t)SEQ * BATCH * 2 * HID;
  float* cn = hn + (size_t)4 * BATCH * HID;

  // ---- workspace layout ----
  char* ws = (char*)d_ws;
  unsigned* counters = (unsigned*)ws;                         // 256 B
  bf16_t* wih0b = (bf16_t*)(ws + 256);                        // 2*2048*512
  bf16_t* wih1b = wih0b + (size_t)2 * GDIM * 512;             // 2*2048*1024
  bf16_t* whhb  = wih1b + (size_t)2 * GDIM * 1024;            // 4*2048*512
  float*  xpbuf = (float*)(whhb + (size_t)4 * GDIM * 512);    // 2*32768*2048 f32
  float*  out0  = xpbuf + (size_t)2 * MROWS * GDIM;           // 512*64*1024 f32
  bf16_t* hmbuf = (bf16_t*)(out0 + (size_t)SEQ * BATCH * 2 * HID); // 2*2*64*512

  hipMemsetAsync(counters, 0, 256, stream);

  // weight conversion f32 -> bf16
  {
    const int n0 = 2 * GDIM * 512, n1 = 2 * GDIM * 1024, n2 = 4 * GDIM * 512;
    cvt_f32_bf16<<<(n0 + 255) / 256, 256, 0, stream>>>(Wih0, wih0b, n0);
    cvt_f32_bf16<<<(n1 + 255) / 256, 256, 0, stream>>>(Wih1, wih1b, n1);
    cvt_f32_bf16<<<(n2 + 255) / 256, 256, 0, stream>>>(Whh, whhb, n2);
  }

  const int gemm_blocks = 2 * (MROWS / 16) * (GDIM / 64) / 8;  // 16384
  const size_t lds_bytes = (size_t)128 * HID * sizeof(bf16_t); // 128 KB

  // layer 0
  xp_gemm<<<gemm_blocks, 256, 0, stream>>>(x, mx0, wih0b, b_ih, b_hh,
                                           xpbuf, 512, 0);
  lstm_scan<<<32, 256, lds_bytes, stream>>>(xpbuf, whhb, mhh, out0, hn, cn,
                                            hmbuf, counters, 0);
  // layer 1 (input = out0, K = 2H = 1024)
  xp_gemm<<<gemm_blocks, 256, 0, stream>>>(out0, mx1, wih1b, b_ih, b_hh,
                                           xpbuf, 1024, 1);
  lstm_scan<<<32, 256, lds_bytes, stream>>>(xpbuf, whhb, mhh, out_main, hn, cn,
                                            hmbuf, counters + 2, 1);

  (void)in_sizes; (void)n_in; (void)out_size; (void)ws_size;
}
